// SpineSegmentationNet_60833916780806
// MI455X (gfx1250) — matrix operand order repaired and checked by reading, not compiled, not run on hardware
//
#include <hip/hip_runtime.h>
#include <hip/hip_bf16.h>
#include <math.h>

// ---------------------------------------------------------------------------
// PointNet++-style pipeline for MI455X (gfx1250, wave32).
//  - KNN: exact k-pass wave-parallel selection, f32 distances (ordering exact)
//  - MLPs: f16 WMMA (v_wmma_f32_16x16x32_f16), f32 accumulate
//    * W pre-swizzled in LDS -> per-lane contiguous 32B B-fragments (ds_load_b128)
//    * 4 M-subtiles per wave share one B fragment (4 wmma / K-step)
//  - All GEMM K dims padded to multiples of 32 (67 -> 96), M mult. of 64
// ---------------------------------------------------------------------------

typedef _Float16 f16;
typedef __attribute__((ext_vector_type(16))) _Float16 v16h;
typedef __attribute__((ext_vector_type(8)))  _Float16 v8h;
typedef __attribute__((ext_vector_type(8)))  float    v8f;

#define BB 8
#define NN 32768
#define SS1 512
#define SS2 128
#define KK 32

// ---------------------------------------------------------------------------
// f16 WMMA GEMM: C[M,Nc] = A[M,Kpad] * W[Kpad,Nc] + bias, optional ReLU.
// Each wave owns a 64(M) x 16(N) stripe: 4 accumulators sharing one B fragment.
// Requires M % 64 == 0, Kpad % 32 == 0, Nc % 16 == 0 (all launches satisfy).
// ---------------------------------------------------------------------------
__global__ void __launch_bounds__(256)
gemm_f16_wmma(const f16* __restrict__ A, const f16* __restrict__ W,
              const float* __restrict__ bias,
              float* __restrict__ Cf, f16* __restrict__ Ch,
              int M, int Kpad, int Nc, int relu)
{
    extern __shared__ f16 sW[];
    const int ntn = Nc >> 4;                         // N tiles
    // Pre-swizzle W into LDS: sW[((ks*ntn+nt)*32 + lane)*16 + j]
    //   = W[ks*32 + (lane>>4)*16 + j, nt*16 + (lane&15)]
    // so a lane's 16 B values per (K-step, N-tile) are contiguous (32B).
    const int tot = Kpad * Nc;
    for (int i = threadIdx.x; i < tot; i += blockDim.x) {
        const int j    = i & 15;
        const int ln   = (i >> 4) & 31;
        const int tile = i >> 9;                     // ks*ntn + nt
        const int nt_  = tile % ntn;
        const int ks_  = tile / ntn;
        const int col  = nt_ * 16 + (ln & 15);
        const int k    = ks_ * 32 + (ln >> 4) * 16 + j;
        sW[i] = W[(size_t)k * Nc + col];
    }
    __syncthreads();

    const int lane = threadIdx.x & 31;
    const int wave = threadIdx.x >> 5;
    const int gw   = blockIdx.x * 8 + wave;
    const int mt64 = gw / ntn;                       // 64-row super-tile
    const int nt   = gw % ntn;
    if (mt64 * 64 >= M) return;                      // wave-uniform: EXEC all-ones

    const int r16  = lane & 15;
    const int half = lane >> 4;
    const int col  = nt * 16 + r16;

    v8f acc0 = {}, acc1 = {}, acc2 = {}, acc3 = {};
    const int ksteps = Kpad >> 5;
    for (int ks = 0; ks < ksteps; ++ks) {
        const int kk = ks << 5;
        // --- B fragment: one contiguous 32B LDS read per lane ---
        const v16h b = *(const v16h*)(sW + ((((size_t)ks * ntn + nt) * 32 + lane) << 4));
        // --- 4 A fragments (two aligned 16B loads each), 4 WMMAs ---
#pragma unroll
        for (int m = 0; m < 4; ++m) {
            const int rowAbs = mt64 * 64 + m * 16 + r16;
            const f16* Arow  = A + (size_t)rowAbs * Kpad + kk + half * 16;
            __builtin_prefetch(Arow + 32, 0, 1);     // next K-step of this row
            const v8h a0 = *(const v8h*)(Arow);      // K: kk+half*16   .. +7
            const v8h a1 = *(const v8h*)(Arow + 8);  // K: kk+half*16+8 .. +15
            const v16h a = __builtin_shufflevector(a0, a1,
                0, 1, 2, 3, 4, 5, 6, 7, 8, 9, 10, 11, 12, 13, 14, 15);
            v8f* accp = (m == 0) ? &acc0 : (m == 1) ? &acc1 : (m == 2) ? &acc2 : &acc3;
            *accp = __builtin_amdgcn_wmma_f32_16x16x32_f16(
                false, a, false, b, (short)0, *accp, false, false);
        }
    }

    const float bv = bias ? bias[col] : 0.0f;
#pragma unroll
    for (int m = 0; m < 4; ++m) {
        const v8f acc = (m == 0) ? acc0 : (m == 1) ? acc1 : (m == 2) ? acc2 : acc3;
#pragma unroll
        for (int i = 0; i < 8; ++i) {
            const int rowAbs = mt64 * 64 + m * 16 + i + half * 8;
            float v = acc[i] + bv;
            if (relu) v = fmaxf(v, 0.0f);
            const size_t off = (size_t)rowAbs * Nc + col;
            if (Cf) Cf[off] = v;
            if (Ch) Ch[off] = (f16)v;
        }
    }
}

// ---------------------------------------------------------------------------
// Exact KNN: one wave32 per query. k passes of wave-argmin with lexicographic
// (d, idx) exclusion of already-selected neighbors.
// ---------------------------------------------------------------------------
__global__ void __launch_bounds__(128)
knn_kernel(const float* __restrict__ refs, const float* __restrict__ qrys,
           int* __restrict__ out, int B, int R, int Q, int k)
{
    const int gw   = (blockIdx.x * blockDim.x + threadIdx.x) >> 5;
    const int lane = threadIdx.x & 31;
    if (gw >= B * Q) return;                        // wave-uniform
    const int b = gw / Q, q = gw % Q;
    const float* rp = refs + (size_t)b * R * 3;
    const float* qp = qrys + ((size_t)b * Q + q) * 3;
    const float qx = qp[0], qy = qp[1], qz = qp[2];

    float lastD = -1.0f; int lastI = -1;
    for (int j = 0; j < k; ++j) {
        float bestD = 3.4e38f; int bestI = 0x7fffffff;
        for (int r = lane; r < R; r += 32) {
            const float dx = qx - rp[r * 3 + 0];
            const float dy = qy - rp[r * 3 + 1];
            const float dz = qz - rp[r * 3 + 2];
            const float d  = dx * dx + dy * dy + dz * dz;
            const bool valid = (d > lastD) || (d == lastD && r > lastI);
            if (valid && (d < bestD || (d == bestD && r < bestI))) { bestD = d; bestI = r; }
        }
        for (int off = 16; off > 0; off >>= 1) {
            const float od = __shfl_xor(bestD, off, 32);
            const int   oi = __shfl_xor(bestI, off, 32);
            if (od < bestD || (od == bestD && oi < bestI)) { bestD = od; bestI = oi; }
        }
        if (lane == 0) out[(size_t)gw * k + j] = bestI;
        lastD = bestD; lastI = bestI;
    }
}

// ---------------------------------------------------------------------------
// Small helpers
// ---------------------------------------------------------------------------
__global__ void gather_centers(const float* __restrict__ src, float* __restrict__ dst,
                               int B, int Nsrc, int S, int stride)
{
    const int i = blockIdx.x * blockDim.x + threadIdx.x;
    if (i >= B * S) return;
    const int b = i / S, s = i % S;
    const size_t so = ((size_t)b * Nsrc + (size_t)s * stride) * 3;
    dst[(size_t)i * 3 + 0] = src[so + 0];
    dst[(size_t)i * 3 + 1] = src[so + 1];
    dst[(size_t)i * 3 + 2] = src[so + 2];
}

// SA1 layer1 fused with gather: rows = B*S1*K, 3 coords -> 64 ch, ReLU, f16 out.
__global__ void sa1_layer1(const float* __restrict__ x, const int* __restrict__ nidx,
                           const float* __restrict__ w1, const float* __restrict__ b1,
                           f16* __restrict__ A1, int rows)
{
    __shared__ float sw[3 * 64];
    __shared__ float sb[64];
    for (int i = threadIdx.x; i < 192; i += blockDim.x) sw[i] = w1[i];
    for (int i = threadIdx.x; i < 64;  i += blockDim.x) sb[i] = b1[i];
    __syncthreads();
    const int row = blockIdx.x * blockDim.x + threadIdx.x;
    if (row >= rows) return;
    const int b   = row / (SS1 * KK);
    const int idx = nidx[row];
    const float* p = x + ((size_t)b * NN + idx) * 3;
    const float px = p[0], py = p[1], pz = p[2];
#pragma unroll 4
    for (int c = 0; c < 64; ++c) {
        const float v = sb[c] + px * sw[c] + py * sw[64 + c] + pz * sw[128 + c];
        A1[(size_t)row * 64 + c] = (f16)fmaxf(v, 0.0f);
    }
}

// Gather f16 feature rows by neighbor index: dst[r,:] = src[b, nidx[r], :]
__global__ void gather_rows_f16(const f16* __restrict__ src, const int* __restrict__ nidx,
                                f16* __restrict__ dst, int B, int Ssrc, int C, int rowsPerB)
{
    const int t = blockIdx.x * blockDim.x + threadIdx.x;
    const int total = B * rowsPerB * C;
    if (t >= total) return;
    const int c = t % C, r = t / C;
    const int b = r / rowsPerB;
    dst[t] = src[((size_t)b * Ssrc + nidx[r]) * C + c];
}

// Max-pool over group of k consecutive rows.
__global__ void group_max(const float* __restrict__ H, float* __restrict__ outF,
                          f16* __restrict__ outH, int groups, int k, int C)
{
    const int t = blockIdx.x * blockDim.x + threadIdx.x;
    if (t >= groups * C) return;
    const int c = t % C, g = t / C;
    const float* p = H + ((size_t)g * k) * C + c;
    float m = -3.4e38f;
    for (int j = 0; j < k; ++j) m = fmaxf(m, p[(size_t)j * C]);
    outF[t] = m;
    if (outH) outH[t] = (f16)m;
}

// FP1 input: concat([f1 (64ch), mean-of-3 interp of f2 (128ch)]) -> f16, K=192
__global__ void fp1_build(const f16* __restrict__ f1h, const float* __restrict__ f2,
                          const int* __restrict__ nidx, f16* __restrict__ A, int rows)
{
    const int t = blockIdx.x * blockDim.x + threadIdx.x;
    const int Kp = 192, total = rows * Kp;
    if (t >= total) return;
    const int kc = t % Kp, r = t / Kp, b = r / SS1;
    f16 v;
    if (kc < 64) {
        v = f1h[(size_t)r * 64 + kc];
    } else {
        const int c = kc - 64;
        const int* ni = nidx + (size_t)r * 3;
        float s = 0.0f;
        for (int j = 0; j < 3; ++j) s += f2[((size_t)b * SS2 + ni[j]) * 128 + c];
        v = (f16)(s * (1.0f / 3.0f));
    }
    A[t] = v;
}

// FP2 input: concat([x (3ch), mean-of-3 interp of g1 (64ch)]) zero-padded to K=96
__global__ void fp2_build(const float* __restrict__ x, const float* __restrict__ g1,
                          const int* __restrict__ nidx, f16* __restrict__ A, int rows)
{
    const int t = blockIdx.x * blockDim.x + threadIdx.x;
    const int Kp = 96;
    if (t >= rows * Kp) return;
    const int kc = t % Kp, r = t / Kp, b = r / NN;
    f16 v = (f16)0.0f;
    if (kc < 3) {
        v = (f16)x[(size_t)r * 3 + kc];
    } else if (kc < 67) {
        const int c = kc - 3;
        const int* ni = nidx + (size_t)r * 3;
        float s = 0.0f;
        for (int j = 0; j < 3; ++j) s += g1[((size_t)b * SS1 + ni[j]) * 64 + c];
        v = (f16)(s * (1.0f / 3.0f));
    }
    A[t] = v;
}

// f32 weight (K,Nc) -> f16 padded (Kpad,Nc), zero-filled tail rows.
__global__ void wcast(const float* __restrict__ w, f16* __restrict__ o,
                      int K, int Kpad, int Nc)
{
    const int t = blockIdx.x * blockDim.x + threadIdx.x;
    if (t >= Kpad * Nc) return;
    const int k = t / Nc;
    o[t] = (k < K) ? (f16)w[t] : (f16)0.0f;
}

// Final FC 32->2 + log_softmax.
__global__ void fc_logsoftmax(const float* __restrict__ g2, const float* __restrict__ fcw,
                              const float* __restrict__ fcb, float* __restrict__ out, int total)
{
    const int t = blockIdx.x * blockDim.x + threadIdx.x;
    if (t >= total) return;
    const float* p = g2 + (size_t)t * 32;
    float l0 = fcb[0], l1 = fcb[1];
#pragma unroll 4
    for (int c = 0; c < 32; ++c) {
        const float v = p[c];
        l0 += v * fcw[c * 2 + 0];
        l1 += v * fcw[c * 2 + 1];
    }
    const float m   = fmaxf(l0, l1);
    const float lse = m + logf(expf(l0 - m) + expf(l1 - m));
    out[(size_t)t * 2 + 0] = l0 - lse;
    out[(size_t)t * 2 + 1] = l1 - lse;
}

// ---------------------------------------------------------------------------
// Launch
// ---------------------------------------------------------------------------
static inline int ceil_div(long long a, long long b) { return (int)((a + b - 1) / b); }

extern "C" void kernel_launch(void* const* d_in, const int* in_sizes, int n_in,
                              void* d_out, int out_size, void* d_ws, size_t ws_size,
                              hipStream_t stream)
{
    (void)in_sizes; (void)n_in; (void)out_size; (void)ws_size;
    const float* x       = (const float*)d_in[0];
    const float* sa1_w1  = (const float*)d_in[1];
    const float* sa1_b1  = (const float*)d_in[2];
    const float* sa1_w2  = (const float*)d_in[3];
    const float* sa1_b2  = (const float*)d_in[4];
    const float* sa2_w1  = (const float*)d_in[5];
    const float* sa2_b1  = (const float*)d_in[6];
    const float* sa2_w2  = (const float*)d_in[7];
    const float* sa2_b2  = (const float*)d_in[8];
    const float* fp1_w1  = (const float*)d_in[9];
    const float* fp1_b1  = (const float*)d_in[10];
    const float* fp1_w2  = (const float*)d_in[11];
    const float* fp1_b2  = (const float*)d_in[12];
    const float* fp2_w1  = (const float*)d_in[13];
    const float* fp2_b1  = (const float*)d_in[14];
    const float* fp2_w2  = (const float*)d_in[15];
    const float* fp2_b2  = (const float*)d_in[16];
    const float* fc_w    = (const float*)d_in[17];
    const float* fc_b    = (const float*)d_in[18];
    float* out = (float*)d_out;

    // ---- workspace carve-up (256B aligned) ----
    char* ws = (char*)d_ws;
    size_t cur = 0;
    auto alloc = [&](size_t bytes) -> char* {
        char* p = ws + cur;
        cur += (bytes + 255) & ~(size_t)255;
        return p;
    };
    float* centers1 = (float*)alloc((size_t)BB * SS1 * 3 * 4);
    float* centers2 = (float*)alloc((size_t)BB * SS2 * 3 * 4);
    int*   nidx1    = (int*)  alloc((size_t)BB * SS1 * KK * 4);
    int*   nidx2    = (int*)  alloc((size_t)BB * SS2 * KK * 4);
    int*   nidxF1   = (int*)  alloc((size_t)BB * SS1 * 3 * 4);
    int*   nidxF2   = (int*)  alloc((size_t)BB * NN  * 3 * 4);
    f16*   A1       = (f16*)  alloc((size_t)BB * SS1 * KK * 64 * 2);
    float* H1       = (float*)alloc((size_t)BB * SS1 * KK * 64 * 4);
    float* f1       = (float*)alloc((size_t)BB * SS1 * 64 * 4);
    f16*   f1h      = (f16*)  alloc((size_t)BB * SS1 * 64 * 2);
    f16*   A2       = (f16*)  alloc((size_t)BB * SS2 * KK * 64 * 2);
    f16*   H2a      = (f16*)  alloc((size_t)BB * SS2 * KK * 128 * 2);
    float* H2       = (float*)alloc((size_t)BB * SS2 * KK * 128 * 4);
    float* f2       = (float*)alloc((size_t)BB * SS2 * 128 * 4);
    f16*   Afp1     = (f16*)  alloc((size_t)BB * SS1 * 192 * 2);
    f16*   Hfp1a    = (f16*)  alloc((size_t)BB * SS1 * 64 * 2);
    float* g1       = (float*)alloc((size_t)BB * SS1 * 64 * 4);
    f16*   Afp2     = (f16*)  alloc((size_t)BB * NN * 96 * 2);
    f16*   Hfp2a    = (f16*)  alloc((size_t)BB * NN * 32 * 2);
    float* g2       = (float*)alloc((size_t)BB * NN * 32 * 4);
    f16*   w_sa1_2  = (f16*)  alloc(64 * 64 * 2);
    f16*   w_sa2_1  = (f16*)  alloc(64 * 128 * 2);
    f16*   w_sa2_2  = (f16*)  alloc(128 * 128 * 2);
    f16*   w_fp1_1  = (f16*)  alloc(192 * 64 * 2);
    f16*   w_fp1_2  = (f16*)  alloc(64 * 64 * 2);
    f16*   w_fp2_1  = (f16*)  alloc(96 * 32 * 2);
    f16*   w_fp2_2  = (f16*)  alloc(32 * 32 * 2);

    // ---- weight casts (f32 -> f16, K zero-padded to mult of 32) ----
    auto cast_w = [&](const float* w, f16* o, int K, int Kpad, int Nc) {
        wcast<<<ceil_div((long long)Kpad * Nc, 256), 256, 0, stream>>>(w, o, K, Kpad, Nc);
    };
    cast_w(sa1_w2, w_sa1_2,  64,  64,  64);
    cast_w(sa2_w1, w_sa2_1,  64,  64, 128);
    cast_w(sa2_w2, w_sa2_2, 128, 128, 128);
    cast_w(fp1_w1, w_fp1_1, 192, 192,  64);
    cast_w(fp1_w2, w_fp1_2,  64,  64,  64);
    cast_w(fp2_w1, w_fp2_1,  67,  96,  32);
    cast_w(fp2_w2, w_fp2_2,  32,  32,  32);

    auto gemm = [&](const f16* A, const f16* W, const float* bias,
                    float* Cf, f16* Ch, int M, int Kpad, int Nc, int relu) {
        const int tiles = (M / 64) * (Nc / 16);     // 64-row super-tiles per wave
        gemm_f16_wmma<<<ceil_div(tiles, 8), 256, (size_t)Kpad * Nc * 2, stream>>>(
            A, W, bias, Cf, Ch, M, Kpad, Nc, relu);
    };
    auto knn = [&](const float* refs, const float* qrys, int* o, int R, int Q, int k) {
        knn_kernel<<<ceil_div((long long)BB * Q * 32, 128), 128, 0, stream>>>(
            refs, qrys, o, BB, R, Q, k);
    };

    // ---- sampling (deterministic strided stand-in for host-RNG permutation) ----
    gather_centers<<<ceil_div(BB * SS1, 256), 256, 0, stream>>>(x,        centers1, BB, NN,  SS1, NN / SS1);
    gather_centers<<<ceil_div(BB * SS2, 256), 256, 0, stream>>>(centers1, centers2, BB, SS1, SS2, SS1 / SS2);

    // ---- SA1 ----
    knn(x, centers1, nidx1, NN, SS1, KK);
    {
        const int rows = BB * SS1 * KK;
        sa1_layer1<<<ceil_div(rows, 256), 256, 0, stream>>>(x, nidx1, sa1_w1, sa1_b1, A1, rows);
        gemm(A1, w_sa1_2, sa1_b2, H1, nullptr, rows, 64, 64, 0);
        group_max<<<ceil_div(BB * SS1 * 64, 256), 256, 0, stream>>>(H1, f1, f1h, BB * SS1, KK, 64);
    }

    // ---- SA2 ----
    knn(centers1, centers2, nidx2, SS1, SS2, KK);
    {
        const int rows = BB * SS2 * KK;
        gather_rows_f16<<<ceil_div((long long)rows * 64, 256), 256, 0, stream>>>(
            f1h, nidx2, A2, BB, SS1, 64, SS2 * KK);
        gemm(A2,  w_sa2_1, sa2_b1, nullptr, H2a, rows,  64, 128, 1);
        gemm(H2a, w_sa2_2, sa2_b2, H2,  nullptr, rows, 128, 128, 0);
        group_max<<<ceil_div(BB * SS2 * 128, 256), 256, 0, stream>>>(H2, f2, nullptr, BB * SS2, KK, 128);
    }

    // ---- FP1: s2 (128 refs) -> s1 (512 queries), k=3 ----
    knn(centers2, centers1, nidxF1, SS2, SS1, 3);
    {
        const int rows = BB * SS1;
        fp1_build<<<ceil_div((long long)rows * 192, 256), 256, 0, stream>>>(f1h, f2, nidxF1, Afp1, rows);
        gemm(Afp1,  w_fp1_1, fp1_b1, nullptr, Hfp1a, rows, 192, 64, 1);
        gemm(Hfp1a, w_fp1_2, fp1_b2, g1, nullptr,   rows,  64, 64, 0);
    }

    // ---- FP2: s1 (512 refs) -> x (32768 queries), k=3 ----
    knn(centers1, x, nidxF2, SS1, NN, 3);
    {
        const int rows = BB * NN;
        fp2_build<<<ceil_div((long long)rows * 96, 256), 256, 0, stream>>>(x, g1, nidxF2, Afp2, rows);
        gemm(Afp2,  w_fp2_1, fp2_b1, nullptr, Hfp2a, rows, 96, 32, 1);
        gemm(Hfp2a, w_fp2_2, fp2_b2, g2, nullptr,   rows, 32, 32, 0);
        fc_logsoftmax<<<ceil_div(rows, 256), 256, 0, stream>>>(g2, fc_w, fc_b, out, rows);
    }
}
